// MambaLayer_6863357739621
// MI455X (gfx1250) — matrix operand order repaired and checked by reading, not compiled, not run on hardware
//
#include <hip/hip_runtime.h>
#include <hip/hip_bf16.h>

// ---------------------------------------------------------------------------
// Mamba2 layer for MI455X (gfx1250): bf16 WMMA projections + register-resident
// selective scan with LDS staging and wave32 shuffles.
// ---------------------------------------------------------------------------

typedef __attribute__((ext_vector_type(16))) __bf16 v16bf;
typedef __attribute__((ext_vector_type(8)))  __bf16 v8bf;
typedef __attribute__((ext_vector_type(8)))  float  v8f;

#define DMODEL   512
#define DINNER   1024
#define DSTATE   128
#define DCONVW   4
#define NBATCH   4
#define SEQL     2048
#define NTOK     (NBATCH * SEQL)          // 8192
#define DINPROJ  2305
#define NPAD     2368                     // 2305 padded up to 64
#define CONVCH   1280                     // D_INNER + 2*D_STATE

// ---- workspace layout (bytes) ----
static constexpr size_t OFF_XBF    = 0;                                   // 8192*512 bf16
static constexpr size_t OFF_WINBF  = OFF_XBF    + (size_t)NTOK*DMODEL*2;  // 2368*512 bf16
static constexpr size_t OFF_WOUTBF = OFF_WINBF  + (size_t)NPAD*DMODEL*2;  // 512*1024 bf16
static constexpr size_t OFF_ZX     = OFF_WOUTBF + (size_t)DMODEL*DINNER*2;// 8192*2368 f32
static constexpr size_t OFF_CONV   = OFF_ZX     + (size_t)NTOK*NPAD*4;    // 8192*1280 f32
static constexpr size_t OFF_DTS    = OFF_CONV   + (size_t)NTOK*CONVCH*4;  // 8192 f32
static constexpr size_t OFF_DAS    = OFF_DTS    + (size_t)NTOK*4;         // 8192 f32
static constexpr size_t OFF_Y      = OFF_DAS    + (size_t)NTOK*4;         // 8192*1024 f32
static constexpr size_t OFF_YNBF   = OFF_Y      + (size_t)NTOK*DINNER*4;  // 8192*1024 bf16

// ---------------------------------------------------------------------------
// float -> bf16 conversion
// ---------------------------------------------------------------------------
__global__ void k_f32_to_bf16(const float* __restrict__ in, __bf16* __restrict__ out, int n) {
    int i = blockIdx.x * 256 + threadIdx.x;
    if (i < n) out[i] = (__bf16)in[i];
}

// W_in (2305,512) -> padded (2368,512) bf16, zero-fill padded rows
__global__ void k_cvt_win(const float* __restrict__ W, __bf16* __restrict__ out) {
    int i = blockIdx.x * 256 + threadIdx.x;
    if (i >= NPAD * DMODEL) return;
    int n = i / DMODEL;
    out[i] = (n < DINPROJ) ? (__bf16)W[i] : (__bf16)0.0f;
}

// ---------------------------------------------------------------------------
// bf16 WMMA GEMM: C[m][n] = sum_k A[m][k] * W[n][k]
// A: (M,K) row-major bf16, W: (N,K) row-major bf16, C: f32 with leading dim ldc.
// Block = 8 waves; wave computes a 16x64 strip; block tile = 128M x 64N.
// gridDim.x = N/64, gridDim.y = M/128.
// Fragment gathers use the same contiguous K order for A and B:
//   A lane(l): m=l&15, run  K in [hi*16, hi*16+16)   (hi=l>>4)
//   B lane(l): n=l&15, run  K in [hi*16, hi*16+16)
// ---------------------------------------------------------------------------
__global__ void k_gemm_bf16(const __bf16* __restrict__ A, const __bf16* __restrict__ W,
                            float* __restrict__ C, int K, int ldc) {
    const int lane = threadIdx.x & 31;
    const int wave = threadIdx.x >> 5;
    const int ln   = lane & 15;
    const int hi   = lane >> 4;
    const int rowA = blockIdx.y * 128 + wave * 16 + ln;
    const int n0   = blockIdx.x * 64;

    const __bf16* arow = A + (size_t)rowA * K;
    const __bf16* b0 = W + (size_t)(n0 +  0 + ln) * K;
    const __bf16* b1 = W + (size_t)(n0 + 16 + ln) * K;
    const __bf16* b2 = W + (size_t)(n0 + 32 + ln) * K;
    const __bf16* b3 = W + (size_t)(n0 + 48 + ln) * K;

    v8f acc0 = {}, acc1 = {}, acc2 = {}, acc3 = {};

    for (int k0 = 0; k0 < K; k0 += 32) {
        // A fragment: two contiguous 16B runs
        v8bf alo = *(const v8bf*)(arow + k0 + hi * 16);
        v8bf ahi = *(const v8bf*)(arow + k0 + hi * 16 + 8);
        v16bf af = __builtin_shufflevector(alo, ahi,
                       0,1,2,3,4,5,6,7,8,9,10,11,12,13,14,15);
        // B fragments: one contiguous 32B run per 16-col sub-tile
        const int bo = k0 + hi * 16;
        v16bf bf0 = __builtin_shufflevector(*(const v8bf*)(b0 + bo), *(const v8bf*)(b0 + bo + 8),
                       0,1,2,3,4,5,6,7,8,9,10,11,12,13,14,15);
        v16bf bf1 = __builtin_shufflevector(*(const v8bf*)(b1 + bo), *(const v8bf*)(b1 + bo + 8),
                       0,1,2,3,4,5,6,7,8,9,10,11,12,13,14,15);
        v16bf bf2 = __builtin_shufflevector(*(const v8bf*)(b2 + bo), *(const v8bf*)(b2 + bo + 8),
                       0,1,2,3,4,5,6,7,8,9,10,11,12,13,14,15);
        v16bf bf3 = __builtin_shufflevector(*(const v8bf*)(b3 + bo), *(const v8bf*)(b3 + bo + 8),
                       0,1,2,3,4,5,6,7,8,9,10,11,12,13,14,15);

        acc0 = __builtin_amdgcn_wmma_f32_16x16x32_bf16(false, af, false, bf0, (short)0, acc0, false, false);
        acc1 = __builtin_amdgcn_wmma_f32_16x16x32_bf16(false, af, false, bf1, (short)0, acc1, false, false);
        acc2 = __builtin_amdgcn_wmma_f32_16x16x32_bf16(false, af, false, bf2, (short)0, acc2, false, false);
        acc3 = __builtin_amdgcn_wmma_f32_16x16x32_bf16(false, af, false, bf3, (short)0, acc3, false, false);
    }

    // C layout: VGPR r, lane l -> row = hi*8 + r, col = l&15
    float* crow = C + (size_t)(blockIdx.y * 128 + wave * 16 + hi * 8) * ldc + n0 + ln;
#pragma unroll
    for (int r = 0; r < 8; ++r) {
        crow[(size_t)r * ldc +  0] = acc0[r];
        crow[(size_t)r * ldc + 16] = acc1[r];
        crow[(size_t)r * ldc + 32] = acc2[r];
        crow[(size_t)r * ldc + 48] = acc3[r];
    }
}

// ---------------------------------------------------------------------------
// dt = softplus(dt_raw + dt_bias), dA = exp(dt * -exp(A_log))   (NHEADS == 1)
// ---------------------------------------------------------------------------
__global__ void k_dt(const float* __restrict__ ZX, const float* __restrict__ dt_bias,
                     const float* __restrict__ A_log, float* __restrict__ dts,
                     float* __restrict__ dAs) {
    int i = blockIdx.x * 256 + threadIdx.x;
    if (i >= NTOK) return;
    float v  = ZX[(size_t)i * NPAD + 2 * DINNER + 2 * DSTATE] + dt_bias[0];
    float sp = (v > 20.f) ? v : log1pf(__expf(v));
    float A  = -__expf(A_log[0]);
    dts[i] = sp;
    dAs[i] = __expf(sp * A);
}

// ---------------------------------------------------------------------------
// causal depthwise conv1d (k=4, left zero-pad) + bias + SiLU over xBC channels
// ---------------------------------------------------------------------------
__global__ void k_conv_silu(const float* __restrict__ ZX, const float* __restrict__ cw,
                            const float* __restrict__ cb, float* __restrict__ out) {
    int i = blockIdx.x * 256 + threadIdx.x;
    if (i >= NTOK * CONVCH) return;
    const int c   = i % CONVCH;
    const int row = i / CONVCH;          // b*2048 + l
    const int l   = row & (SEQL - 1);
    float acc = cb[c];
#pragma unroll
    for (int j = 0; j < DCONVW; ++j) {
        int lj = l - (DCONVW - 1) + j;
        if (lj >= 0)
            acc += cw[c * DCONVW + j] *
                   ZX[(size_t)(row - (DCONVW - 1) + j) * NPAD + DINNER + c];
    }
    out[i] = acc / (1.f + __expf(-acc));  // silu
}

// ---------------------------------------------------------------------------
// Selective scan. grid = (16 P-tiles, 4 batches), 256 threads.
// Thread owns p = p0 + (tid>>2), n-chunk = (tid&3)*32..+32; 32 f32 state regs.
// B_t / C_t / x_t staged in LDS each step; y reduced via wave32 shfl_xor.
// ---------------------------------------------------------------------------
__global__ void k_scan(const float* __restrict__ conv, const float* __restrict__ dts,
                       const float* __restrict__ dAs, const float* __restrict__ Dp,
                       float* __restrict__ Y) {
    __shared__ float sB[DSTATE];
    __shared__ float sC[DSTATE];
    __shared__ float sX[64];
    const int tid = threadIdx.x;
    const int b   = blockIdx.y;
    const int p0  = blockIdx.x * 64;
    const int pl  = tid >> 2;
    const int nq  = tid & 3;
    const float Dp0 = Dp[0];

    float st[32];
#pragma unroll
    for (int i = 0; i < 32; ++i) st[i] = 0.f;

    for (int t = 0; t < SEQL; ++t) {
        const size_t base = ((size_t)b * SEQL + t) * CONVCH;
        if (tid < 128) sB[tid]       = conv[base + DINNER + tid];
        else           sC[tid - 128] = conv[base + DINNER + DSTATE + (tid - 128)];
        if (tid < 64)  sX[tid]       = conv[base + p0 + tid];
        if (t + 8 < SEQL) {   // hide HBM latency across the serial chain
            const size_t nb = ((size_t)b * SEQL + (t + 8)) * CONVCH;
            if (tid < 64) __builtin_prefetch(&conv[nb + DINNER + tid * 4], 0, 0);
            if (tid < 16) __builtin_prefetch(&conv[nb + p0 + tid * 4], 0, 0);
        }
        __syncthreads();

        const size_t ti  = (size_t)b * SEQL + t;
        const float dAv  = dAs[ti];
        const float dtv  = dts[ti];
        const float xv   = sX[pl];
        const float coef = dtv * xv;
        const float* Bp  = &sB[nq * 32];
        const float* Cp  = &sC[nq * 32];

        float acc = 0.f;
#pragma unroll
        for (int i = 0; i < 32; ++i) {
            st[i] = dAv * st[i] + coef * Bp[i];
            acc  += st[i] * Cp[i];
        }
        acc += __shfl_xor(acc, 1);
        acc += __shfl_xor(acc, 2);
        if (nq == 0)
            Y[ti * (size_t)DINNER + p0 + pl] = acc + Dp0 * xv;
        __syncthreads();
    }
}

// ---------------------------------------------------------------------------
// gated RMSNorm: g = y * silu(z); out = g * rsqrt(mean(g^2)+eps) * w  -> bf16
// one block per token, 4 channels/thread
// ---------------------------------------------------------------------------
__global__ void k_gated_rmsnorm(const float* __restrict__ Y, const float* __restrict__ ZX,
                                const float* __restrict__ nw, __bf16* __restrict__ out) {
    __shared__ float red[8];
    const int row = blockIdx.x;
    const int tid = threadIdx.x;
    const float* yrow = Y  + (size_t)row * DINNER;
    const float* zrow = ZX + (size_t)row * NPAD;   // z = cols [0,1024)
    float g[4];
    float ss = 0.f;
#pragma unroll
    for (int j = 0; j < 4; ++j) {
        const int c = tid * 4 + j;
        float z = zrow[c];
        float s = z / (1.f + __expf(-z));
        g[j] = yrow[c] * s;
        ss += g[j] * g[j];
    }
#pragma unroll
    for (int o = 16; o >= 1; o >>= 1) ss += __shfl_xor(ss, o);
    if ((tid & 31) == 0) red[tid >> 5] = ss;
    __syncthreads();
    float tot = red[0] + red[1] + red[2] + red[3] + red[4] + red[5] + red[6] + red[7];
    const float scale = rsqrtf(tot * (1.f / (float)DINNER) + 1e-5f);
#pragma unroll
    for (int j = 0; j < 4; ++j) {
        const int c = tid * 4 + j;
        out[(size_t)row * DINNER + c] = (__bf16)(g[j] * scale * nw[c]);
    }
}

__global__ void k_copy_f32(const float* __restrict__ in, float* __restrict__ out, int n) {
    int i = blockIdx.x * 256 + threadIdx.x;
    if (i < n) out[i] = in[i];
}

// ---------------------------------------------------------------------------
extern "C" void kernel_launch(void* const* d_in, const int* in_sizes, int n_in,
                              void* d_out, int out_size, void* d_ws, size_t ws_size,
                              hipStream_t stream) {
    const float* x         = (const float*)d_in[0];
    const float* rnn_state = (const float*)d_in[1];
    const float* W_in      = (const float*)d_in[2];
    const float* conv_w    = (const float*)d_in[3];
    const float* conv_b    = (const float*)d_in[4];
    const float* dt_bias   = (const float*)d_in[5];
    const float* A_log     = (const float*)d_in[6];
    const float* Dp        = (const float*)d_in[7];
    const float* norm_w    = (const float*)d_in[8];
    const float* W_out     = (const float*)d_in[9];

    char* ws = (char*)d_ws;
    __bf16* xbf    = (__bf16*)(ws + OFF_XBF);
    __bf16* winbf  = (__bf16*)(ws + OFF_WINBF);
    __bf16* woutbf = (__bf16*)(ws + OFF_WOUTBF);
    float*  zx     = (float*)(ws + OFF_ZX);
    float*  conv   = (float*)(ws + OFF_CONV);
    float*  dts    = (float*)(ws + OFF_DTS);
    float*  dAs    = (float*)(ws + OFF_DAS);
    float*  y      = (float*)(ws + OFF_Y);
    __bf16* ynbf   = (__bf16*)(ws + OFF_YNBF);

    float* out       = (float*)d_out;                       // (B,L,512)
    float* out_state = out + (size_t)NTOK * DMODEL;         // passthrough rnn_state

    // 1) bf16 conversions for WMMA operands
    k_f32_to_bf16<<<(NTOK * DMODEL + 255) / 256, 256, 0, stream>>>(x, xbf, NTOK * DMODEL);
    k_cvt_win<<<(NPAD * DMODEL + 255) / 256, 256, 0, stream>>>(W_in, winbf);
    k_f32_to_bf16<<<(DMODEL * DINNER + 255) / 256, 256, 0, stream>>>(W_out, woutbf, DMODEL * DINNER);

    // 2) in_proj GEMM: (8192x512)x(512x2368) -> zx (ld 2368)
    k_gemm_bf16<<<dim3(NPAD / 64, NTOK / 128), 256, 0, stream>>>(xbf, winbf, zx, DMODEL, NPAD);

    // 3) dt / dA precompute
    k_dt<<<(NTOK + 255) / 256, 256, 0, stream>>>(zx, dt_bias, A_log, dts, dAs);

    // 4) depthwise conv + silu over xBC
    k_conv_silu<<<(NTOK * CONVCH + 255) / 256, 256, 0, stream>>>(zx, conv_w, conv_b, conv);

    // 5) selective scan -> y (includes + Dp*xs)
    k_scan<<<dim3(DINNER / 64, NBATCH), 256, 0, stream>>>(conv, dts, dAs, Dp, y);

    // 6) gated RMSNorm -> bf16
    k_gated_rmsnorm<<<NTOK, 256, 0, stream>>>(y, zx, norm_w, ynbf);

    // 7) out_proj GEMM: (8192x1024)x(1024x512) -> d_out
    k_gemm_bf16<<<dim3(DMODEL / 64, NTOK / 128), 256, 0, stream>>>(ynbf, woutbf, out, DINNER, DMODEL);

    // 8) rnn_state passthrough
    k_copy_f32<<<(NBATCH * DMODEL + 255) / 256, 256, 0, stream>>>(rnn_state, out_state, NBATCH * DMODEL);
}
